// RelPartialLearnableMultiHeadAttn_64914135712023
// MI455X (gfx1250) — matrix operand order repaired and checked by reading, not compiled, not run on hardware
//
#include <hip/hip_runtime.h>
#include <hip/hip_bf16.h>
#include <math.h>

typedef __bf16 bf16_t;
typedef __attribute__((ext_vector_type(16))) __bf16 v16bf;
typedef __attribute__((ext_vector_type(8)))  __bf16 v8bf;
typedef __attribute__((ext_vector_type(8)))  float  v8f;
typedef unsigned int u32;
typedef __attribute__((ext_vector_type(4))) unsigned int u32x4;
typedef __attribute__((ext_vector_type(8))) int i32x8;
typedef __attribute__((ext_vector_type(4))) int i32x4;

#define QLEN   2048
#define BATCH  2
#define DMODEL 1024
#define NH     16
#define DHEAD  64
#define SCALE  0.125f
#define NEGINF (-3.0e38f)

#if defined(__HIP_DEVICE_COMPILE__) && __has_builtin(__builtin_amdgcn_tensor_load_to_lds)
#define USE_TDM 1
#else
#define USE_TDM 0
#endif

union FragU { v16bf v; v8bf h[2]; };

__device__ __forceinline__ v8f zero8f() {
  v8f z = {0.f, 0.f, 0.f, 0.f, 0.f, 0.f, 0.f, 0.f};
  return z;
}

__device__ __forceinline__ v8f wmma_bf16f32(v16bf a, v16bf b, v8f c) {
  // D = A(16x32 bf16) * B(32x16 bf16) + C(16x16 f32)
  return __builtin_amdgcn_wmma_f32_16x16x32_bf16(false, a, false, b, (short)0, c,
                                                 false, false);
}

// A-operand (16x32 bf16) fragment. Caller passes pointer to THIS LANE's row
// start (row = lane&15). Layout: lanes 0-15 hold K 0..15,
// lanes 16-31 hold K 16..31 (consecutive, matching the B fragment).
__device__ __forceinline__ v16bf frag_a_row(const bf16_t* rowptr) {
  const int lane = threadIdx.x & 31;
  const int o = (lane < 16) ? 0 : 16;
  FragU f;
  f.h[0] = *reinterpret_cast<const v8bf*>(rowptr + o);
  f.h[1] = *reinterpret_cast<const v8bf*>(rowptr + 8 + o);
  return f.v;
}

// B-operand (32x16 bf16) fragment. Layout used here: lane's column N = lane&15;
// lanes 0-15 hold K 0..15, lanes 16-31 hold K 16..31 (consecutive, packed).
// Caller passes pointer to this lane's first K element (16 consecutive bf16).
__device__ __forceinline__ v16bf frag_b_krun(const bf16_t* p) {
  FragU f;
  f.h[0] = *reinterpret_cast<const v8bf*>(p);
  f.h[1] = *reinterpret_cast<const v8bf*>(p + 8);
  return f.v;
}

#if USE_TDM
// Issue a TDM 2-D tile load (tile_d0 x tile_d1 bf16, row stride = tensor_d0
// elements) from global to LDS byte offset `lds_byte`. D# packing per CDNA5
// ISA 8.3/8.4: group0 {count=1, lds_addr, global_addr, type=2},
// group1 {data_size=1(2B), tensor dims, tile dims, dim0 stride}.
__device__ __forceinline__ void tdm_load_2d(u32 lds_byte, const bf16_t* gp,
                                            int tensor_d0, int tensor_d1,
                                            int tile_d0, int tile_d1) {
  const unsigned long long ga = (unsigned long long)(size_t)gp;
  u32x4 g0;
  g0[0] = 1u;                                            // count=1 (user D#)
  g0[1] = lds_byte;                                      // lds_addr
  g0[2] = (u32)(ga & 0xffffffffu);                       // global_addr lo
  g0[3] = (u32)((ga >> 32) & 0x01ffffffu) | (2u << 30);  // addr[56:32]|type=2
  i32x8 g1;
  g1[0] = (int)(1u << 16);                               // data_size=1 (2B)
  g1[1] = (int)(((u32)tensor_d0 & 0xffffu) << 16);       // tensor_dim0 lo16
  g1[2] = (int)((((u32)tensor_d0 >> 16) & 0xffffu) |
                (((u32)tensor_d1 & 0xffffu) << 16));     // dim0 hi | dim1 lo
  g1[3] = (int)((((u32)tensor_d1 >> 16) & 0xffffu) |
                ((u32)tile_d0 << 16));                   // dim1 hi | tile_dim0
  g1[4] = (int)((u32)tile_d1);                           // tile_dim1 (dim2=0)
  g1[5] = tensor_d0;                                     // dim0 stride lo32
  g1[6] = 0;
  g1[7] = 0;
  i32x4 gz = {0, 0, 0, 0};
#if __clang_major__ >= 23
  i32x8 gz8 = {0, 0, 0, 0, 0, 0, 0, 0};
  __builtin_amdgcn_tensor_load_to_lds(g0, g1, gz, gz, gz8, 0);
#else
  __builtin_amdgcn_tensor_load_to_lds(g0, g1, gz, gz, 0);
#endif
}
#endif

// ---------------------------------------------------------------- convert
struct __align__(8) bf16x4 { bf16_t a, b, c, d; };

__global__ void k_cvt(const float4* __restrict__ src, bf16x4* __restrict__ dst,
                      int n4) {
  int i = blockIdx.x * blockDim.x + threadIdx.x;
  if (i < n4) {
    float4 f = src[i];
    bf16x4 o;
    o.a = (bf16_t)f.x; o.b = (bf16_t)f.y; o.c = (bf16_t)f.z; o.d = (bf16_t)f.w;
    dst[i] = o;
  }
}

// ---------------------------------------------------------------- GEMM
// C[M][N] = A[M][K] * W[N][K]^T  (bf16 in, f32 accum), M,N multiples of 64,
// K multiple of 32. 128 threads = 4 waves; block tile 64x64; wave tile 16x64.
// K-tiles staged to LDS by the Tensor Data Mover (double buffered, wave 0
// drives the DMA, s_wait_tensorcnt + barrier for hand-off).
// EPI 0: QKV epilogue (q+bias scatter / k / v-transposed bf16 tensors)
// EPI 1: r-projection epilogue (store reversed rk_rev bf16)
// EPI 2: plain f32 store
template <int EPI>
__launch_bounds__(128)
__global__ void k_gemm(const bf16_t* __restrict__ A, const bf16_t* __restrict__ W,
                       int K, int N, float* __restrict__ Cf,
                       bf16_t* __restrict__ qac, bf16_t* __restrict__ qbd,
                       bf16_t* __restrict__ wkk, bf16_t* __restrict__ wvt,
                       const float* __restrict__ rwb, const float* __restrict__ rrb,
                       bf16_t* __restrict__ rkrev) {
  // [buf][0]=A-tile, [buf][1]=W-tile; single shared object => LDS offset 0.
  __shared__ bf16_t tiles[2][2][64][32];
  const int tid = threadIdx.x, lane = tid & 31, wave = tid >> 5;
  const int m0 = blockIdx.x * 64, n0 = blockIdx.y * 64;
  const int nl = lane & 15, rs = (lane >= 16) ? 8 : 0;

  v8f acc[4];
#pragma unroll
  for (int nt = 0; nt < 4; ++nt) acc[nt] = zero8f();

#if USE_TDM
  if (wave == 0) {
    tdm_load_2d(0 * 4096, A + (size_t)m0 * K, K, 64, 32, 64);
    tdm_load_2d(1 * 4096, W + (size_t)n0 * K, K, 64, 32, 64);
    __builtin_amdgcn_s_wait_tensorcnt(0);
  }
  __syncthreads();
  for (int k0 = 0; k0 < K; k0 += 32) {
    const int buf = (k0 >> 5) & 1;
    const bool pre = (k0 + 32) < K;
    if (wave == 0 && pre) {
      const u32 lb = (u32)((buf ^ 1) * 2 * 4096);
      tdm_load_2d(lb, A + (size_t)m0 * K + k0 + 32, K, 64, 32, 64);
      tdm_load_2d(lb + 4096, W + (size_t)n0 * K + k0 + 32, K, 64, 32, 64);
    }
    v16bf af = frag_a_row(&tiles[buf][0][wave * 16 + nl][0]);
#pragma unroll
    for (int nt = 0; nt < 4; ++nt) {
      v16bf bf = frag_b_krun(&tiles[buf][1][nt * 16 + nl][(lane < 16) ? 0 : 16]);
      acc[nt] = wmma_bf16f32(af, bf, acc[nt]);
    }
    if (wave == 0 && pre) __builtin_amdgcn_s_wait_tensorcnt(0);
    __syncthreads();
  }
#else
  const int arow = tid >> 1, ahalf = (tid & 1) * 16;
  for (int k0 = 0; k0 < K; k0 += 32) {
    const uint4* ga =
        reinterpret_cast<const uint4*>(A + (size_t)(m0 + arow) * K + k0 + ahalf);
    uint4* la = reinterpret_cast<uint4*>(&tiles[0][0][arow][ahalf]);
    la[0] = ga[0]; la[1] = ga[1];
    const uint4* gw =
        reinterpret_cast<const uint4*>(W + (size_t)(n0 + arow) * K + k0 + ahalf);
    uint4* lw = reinterpret_cast<uint4*>(&tiles[0][1][arow][ahalf]);
    lw[0] = gw[0]; lw[1] = gw[1];
    __syncthreads();
    v16bf af = frag_a_row(&tiles[0][0][wave * 16 + nl][0]);
#pragma unroll
    for (int nt = 0; nt < 4; ++nt) {
      v16bf bf = frag_b_krun(&tiles[0][1][nt * 16 + nl][(lane < 16) ? 0 : 16]);
      acc[nt] = wmma_bf16f32(af, bf, acc[nt]);
    }
    __syncthreads();
  }
#endif

  // C layout: lane column N = lane&15; VGPR v -> row M = v + 8*(lane>=16)
#pragma unroll
  for (int nt = 0; nt < 4; ++nt) {
#pragma unroll
    for (int v = 0; v < 8; ++v) {
      const int m = m0 + wave * 16 + rs + v;
      const int n = n0 + nt * 16 + nl;
      const float c = acc[nt][v];
      if (EPI == 0) {
        const int q = m >> 1, b = m & 1;            // rows are q*B + b
        const int sec = n >> 10, hd = n & 1023;     // 0:q 1:k 2:v
        const int h = hd >> 6, dh = hd & 63;
        if (sec == 0) {
          const size_t idx = ((size_t)(b * NH + h) * QLEN + q) * DHEAD + dh;
          qac[idx] = (bf16_t)(c + rwb[hd]);
          qbd[idx] = (bf16_t)(c + rrb[hd]);
        } else if (sec == 1) {
          wkk[((size_t)(b * NH + h) * QLEN + q) * DHEAD + dh] = (bf16_t)c;
        } else {
          // V stored transposed: wvt[b][h][dh][q] so the PV WMMA B-operand
          // (16 consecutive keys per lane at fixed dh) is a contiguous load.
          wvt[((size_t)(b * NH + h) * DHEAD + dh) * QLEN + q] = (bf16_t)c;
        }
      } else if (EPI == 1) {
        const int h = n >> 6, dh = n & 63;
        // store reversed: rk_rev[h][d] = r_head_k[h][Q-1-d]
        rkrev[((size_t)h * QLEN + (size_t)(QLEN - 1 - m)) * DHEAD + dh] =
            (bf16_t)c;
      } else {
        Cf[(size_t)m * N + n] = c;
      }
    }
  }
}

// ---------------------------------------------------------------- attention
// Flash-style causal attention per (b,h). Block = 4 waves, each wave owns a
// 16-row query tile. Score(i,j) = (q_ac[i].k[j] + G[i, i-j]) * scale, where
// G = q_bd . rk_rev^T  (rel-shift trick: BD_shift[i,j] = BDfull[i,Q-1-(i-j)]).
__launch_bounds__(128)
__global__ void k_attn(const bf16_t* __restrict__ qac, const bf16_t* __restrict__ qbd,
                       const bf16_t* __restrict__ wk, const bf16_t* __restrict__ wvt,
                       const bf16_t* __restrict__ rkrev, bf16_t* __restrict__ vec) {
  __shared__ float  Gs[4][16][48];   // per-wave slab of G, cols d0..d0+47
  __shared__ bf16_t Ps[4][16][32];   // per-wave prob tile (for A-layout repack)

  const int tid = threadIdx.x, lane = tid & 31, wave = tid >> 5;
  const int nl = lane & 15, rs = (lane >= 16) ? 8 : 0;
  const int bh = blockIdx.y;                 // b*NH + h
  const int i0 = blockIdx.x * 64 + wave * 16;
  const size_t headbase = (size_t)bh * QLEN * DHEAD;
  const size_t rkbase = (size_t)(bh & (NH - 1)) * QLEN * DHEAD;
  const int kb16 = (lane < 16) ? 0 : 16;

  // Q fragments (A operand row = i0 + (lane&15)), 2 K-chunks covering DH=64
  const bf16_t* qrow_ac = qac + headbase + (size_t)(i0 + nl) * DHEAD;
  const bf16_t* qrow_bd = qbd + headbase + (size_t)(i0 + nl) * DHEAD;
  v16bf qa[2], qb[2];
#pragma unroll
  for (int c = 0; c < 2; ++c) {
    qa[c] = frag_a_row(qrow_ac + 32 * c);
    qb[c] = frag_a_row(qrow_bd + 32 * c);
  }

  v8f oacc[4];
#pragma unroll
  for (int c = 0; c < 4; ++c) oacc[c] = zero8f();
  float mrow[8], lrow[8];
#pragma unroll
  for (int v = 0; v < 8; ++v) { mrow[v] = NEGINF; lrow[v] = 0.0f; }

  for (int jb = 0; jb <= i0 + 15; jb += 32) {
    // ---- G slab: d = i - j ranges over [d0, d0+47] for this key block ----
    const int d0 = i0 - jb - 31;
#pragma unroll
    for (int g = 0; g < 3; ++g) {
      const int rr = d0 + g * 16 + nl;       // rk_rev row for this lane's col
      v8f gacc = zero8f();
#pragma unroll
      for (int c = 0; c < 2; ++c) {
        v16bf bfv;
        if (rr >= 0 && rr < QLEN) {
          bfv = frag_b_krun(rkrev + rkbase + (size_t)rr * DHEAD + 32 * c + kb16);
        } else {
#pragma unroll
          for (int e = 0; e < 16; ++e) bfv[e] = (__bf16)0.0f;
        }
        gacc = wmma_bf16f32(qb[c], bfv, gacc);
      }
#pragma unroll
      for (int v = 0; v < 8; ++v) Gs[wave][rs + v][g * 16 + nl] = gacc[v];
    }

    // ---- scores: AC via WMMA + BD diagonal gather from Gs ----
    float sc[2][8];
    float mnew[8];
#pragma unroll
    for (int v = 0; v < 8; ++v) mnew[v] = NEGINF;

#pragma unroll
    for (int t = 0; t < 2; ++t) {
      const int jt = jb + 16 * t;
      v8f s = zero8f();
      const bf16_t* kp = wk + headbase + (size_t)(jt + nl) * DHEAD;
#pragma unroll
      for (int c = 0; c < 2; ++c) {
        v16bf bfv = frag_b_krun(kp + 32 * c + kb16);
        s = wmma_bf16f32(qa[c], bfv, s);
      }
#pragma unroll
      for (int v = 0; v < 8; ++v) {
        const int M = rs + v;
        const int i = i0 + M;
        const int j = jt + nl;
        const int gc = M - (nl + 16 * t) + 31;    // (i-j) - d0, in [0,46]
        float val = (s[v] + Gs[wave][M][gc]) * SCALE;
        if (j > i) val = NEGINF;                   // causal mask
        sc[t][v] = val;
        mnew[v] = fmaxf(mnew[v], val);
      }
    }

    // ---- online softmax update (row reduce across the 16-lane half) ----
    float pv[2][8];
#pragma unroll
    for (int v = 0; v < 8; ++v) {
      float m = mnew[v];
#pragma unroll
      for (int d = 1; d < 16; d <<= 1) m = fmaxf(m, __shfl_xor(m, d, 16));
      const float mc = fmaxf(mrow[v], m);
      const float alpha = __expf(mrow[v] - mc);
      mrow[v] = mc;
      float rsum = 0.f;
#pragma unroll
      for (int t = 0; t < 2; ++t) {
        const float p = __expf(sc[t][v] - mc);
        pv[t][v] = p;
        rsum += p;
      }
#pragma unroll
      for (int d = 1; d < 16; d <<= 1) rsum += __shfl_xor(rsum, d, 16);
      lrow[v] = lrow[v] * alpha + rsum;
#pragma unroll
      for (int c = 0; c < 4; ++c) oacc[c][v] *= alpha;
    }

    // ---- repack P (C layout -> A layout) via per-wave LDS slab ----
#pragma unroll
    for (int t = 0; t < 2; ++t)
#pragma unroll
      for (int v = 0; v < 8; ++v)
        Ps[wave][rs + v][16 * t + nl] = (bf16_t)pv[t][v];

    v16bf pf = frag_a_row(&Ps[wave][nl][0]);

    // ---- O += P x V (4 dh-column tiles, V pre-transposed: contiguous) ----
#pragma unroll
    for (int c = 0; c < 4; ++c) {
      const bf16_t* vp =
          wvt + ((size_t)bh * DHEAD + 16 * c + nl) * QLEN + jb + kb16;
      oacc[c] = wmma_bf16f32(pf, frag_b_krun(vp), oacc[c]);
    }
  }

  // ---- normalize and write vec[q*B+b][h*DH + dh] (bf16) ----
  const int b = bh >> 4, h = bh & (NH - 1);
#pragma unroll
  for (int v = 0; v < 8; ++v) {
    const float inv = 1.0f / lrow[v];
    const int q = i0 + rs + v;
    bf16_t* op = vec + ((size_t)q * BATCH + b) * (NH * DHEAD) + h * DHEAD;
#pragma unroll
    for (int c = 0; c < 4; ++c)
      op[16 * c + nl] = (bf16_t)(oacc[c][v] * inv);
  }
}

// ---------------------------------------------------------------- layernorm
__launch_bounds__(256)
__global__ void k_ln(const float* __restrict__ w, const float* __restrict__ attn,
                     const float* __restrict__ g, const float* __restrict__ bta,
                     float* __restrict__ out) {
  const int m = blockIdx.x, tid = threadIdx.x;
  __shared__ float red[2][8];
  float x[4];
  float s = 0.f;
#pragma unroll
  for (int e = 0; e < 4; ++e) {
    const int d = tid + 256 * e;
    x[e] = w[(size_t)m * DMODEL + d] + attn[(size_t)m * DMODEL + d];
    s += x[e];
  }
#pragma unroll
  for (int d = 1; d < 32; d <<= 1) s += __shfl_xor(s, d, 32);
  if ((tid & 31) == 0) red[0][tid >> 5] = s;
  __syncthreads();
  float mean = 0.f;
#pragma unroll
  for (int wv = 0; wv < 8; ++wv) mean += red[0][wv];
  mean *= (1.0f / DMODEL);
  float vs = 0.f;
#pragma unroll
  for (int e = 0; e < 4; ++e) { const float dd = x[e] - mean; vs += dd * dd; }
#pragma unroll
  for (int d = 1; d < 32; d <<= 1) vs += __shfl_xor(vs, d, 32);
  if ((tid & 31) == 0) red[1][tid >> 5] = vs;
  __syncthreads();
  float var = 0.f;
#pragma unroll
  for (int wv = 0; wv < 8; ++wv) var += red[1][wv];
  var *= (1.0f / DMODEL);
  const float rstd = rsqrtf(var + 1e-5f);
#pragma unroll
  for (int e = 0; e < 4; ++e) {
    const int d = tid + 256 * e;
    out[(size_t)m * DMODEL + d] = (x[e] - mean) * rstd * g[d] + bta[d];
  }
}

// ---------------------------------------------------------------- launcher
extern "C" void kernel_launch(void* const* d_in, const int* in_sizes, int n_in,
                              void* d_out, int out_size, void* d_ws, size_t ws_size,
                              hipStream_t stream) {
  (void)in_sizes; (void)n_in; (void)out_size; (void)ws_size;
  const float* w     = (const float*)d_in[0];
  const float* r     = (const float*)d_in[1];
  const float* rwb   = (const float*)d_in[2];
  const float* rrb   = (const float*)d_in[3];
  const float* qkvw  = (const float*)d_in[4];
  const float* rnetw = (const float*)d_in[5];
  const float* ow    = (const float*)d_in[6];
  const float* lnw   = (const float*)d_in[7];
  const float* lnb   = (const float*)d_in[8];
  float* out = (float*)d_out;

  char* ws = (char*)d_ws;
  size_t off = 0;
  auto alloc = [&](size_t bytes) -> void* {
    void* p = ws + off;
    off = (off + bytes + 255) & ~(size_t)255;
    return p;
  };
  const size_t MB = QLEN * BATCH;  // 4096 rows
  bf16_t* wbf   = (bf16_t*)alloc(MB * DMODEL * 2);
  bf16_t* rbf   = (bf16_t*)alloc((size_t)QLEN * DMODEL * 2);
  bf16_t* qkvbf = (bf16_t*)alloc((size_t)3 * DMODEL * DMODEL * 2);
  bf16_t* rnbf  = (bf16_t*)alloc((size_t)DMODEL * DMODEL * 2);
  bf16_t* owbf  = (bf16_t*)alloc((size_t)DMODEL * DMODEL * 2);
  bf16_t* qac   = (bf16_t*)alloc((size_t)BATCH * NH * QLEN * DHEAD * 2);
  bf16_t* qbd   = (bf16_t*)alloc((size_t)BATCH * NH * QLEN * DHEAD * 2);
  bf16_t* wkk   = (bf16_t*)alloc((size_t)BATCH * NH * QLEN * DHEAD * 2);
  bf16_t* wvt   = (bf16_t*)alloc((size_t)BATCH * NH * QLEN * DHEAD * 2);
  bf16_t* rkrev = (bf16_t*)alloc((size_t)NH * QLEN * DHEAD * 2);
  bf16_t* vec   = (bf16_t*)alloc(MB * DMODEL * 2);
  float*  attnf = (float*)alloc(MB * DMODEL * 4);

  auto cvt = [&](const float* src, bf16_t* dst, size_t n) {
    const int n4 = (int)(n / 4);
    k_cvt<<<dim3((n4 + 255) / 256), dim3(256), 0, stream>>>(
        (const float4*)src, (bf16x4*)dst, n4);
  };
  cvt(w, wbf, MB * DMODEL);
  cvt(r, rbf, (size_t)QLEN * DMODEL);
  cvt(qkvw, qkvbf, (size_t)3 * DMODEL * DMODEL);
  cvt(rnetw, rnbf, (size_t)DMODEL * DMODEL);
  cvt(ow, owbf, (size_t)DMODEL * DMODEL);

  const dim3 blk(128);
  // QKV projection + per-head scatter with r_w/r_r bias fusion, V transposed
  k_gemm<0><<<dim3(MB / 64, (3 * DMODEL) / 64), blk, 0, stream>>>(
      wbf, qkvbf, DMODEL, 3 * DMODEL, nullptr, qac, qbd, wkk, wvt, rwb, rrb,
      nullptr);
  // r projection, stored reversed per head
  k_gemm<1><<<dim3(QLEN / 64, DMODEL / 64), blk, 0, stream>>>(
      rbf, rnbf, DMODEL, DMODEL, nullptr, nullptr, nullptr, nullptr, nullptr,
      nullptr, nullptr, rkrev);
  // fused causal rel-attention
  k_attn<<<dim3(QLEN / 64, BATCH * NH), blk, 0, stream>>>(qac, qbd, wkk, wvt,
                                                          rkrev, vec);
  // output projection
  k_gemm<2><<<dim3(MB / 64, DMODEL / 64), blk, 0, stream>>>(
      vec, owbf, DMODEL, DMODEL, attnf, nullptr, nullptr, nullptr, nullptr,
      nullptr, nullptr, nullptr);
  // residual + layernorm
  k_ln<<<dim3((unsigned)MB), dim3(256), 0, stream>>>(w, attnf, lnw, lnb, out);
}
